// FP8DbrxExpertGLU_37400575213967
// MI455X (gfx1250) — 8-bit matrix operand order repaired and checked by reading, compiled, not run on hardware
//
#include <hip/hip_runtime.h>
#include <hip/hip_bf16.h>

typedef __attribute__((ext_vector_type(16))) int   v16i;
typedef __attribute__((ext_vector_type(8)))  float v8f;

#if defined(__has_builtin)
# if __has_builtin(__builtin_amdgcn_global_load_async_to_lds_b128)
#  define HAVE_ASYNC_LDS 1
# endif
#endif

#if defined(HAVE_ASYNC_LDS)
typedef int v4i_vec __attribute__((vector_size(16)));
typedef __attribute__((address_space(1))) v4i_vec as1_v4i;
typedef __attribute__((address_space(3))) v4i_vec as3_v4i;
__device__ __forceinline__ void async_cp16(const unsigned char* g, unsigned char* l) {
    __builtin_amdgcn_global_load_async_to_lds_b128(
        (as1_v4i*)(unsigned long long)g,
        (as3_v4i*)(unsigned int)(unsigned long long)l,
        0, 0);
}
# if __has_builtin(__builtin_amdgcn_s_wait_asynccnt)
#  define WAIT_ASYNC(n) __builtin_amdgcn_s_wait_asynccnt(n)
# else
#  define WAIT_ASYNC(n) asm volatile("s_wait_asynccnt %0" :: "i"(n) : "memory")
# endif
#endif

__device__ __forceinline__ unsigned int f32_to_e4m3(float f) {
    unsigned int x    = __float_as_uint(f);
    unsigned int sign = (x >> 24) & 0x80u;
    unsigned int ax   = x & 0x7FFFFFFFu;
    if (ax >= 0x43E00000u) return sign | 0x7Eu;
    if (ax <  0x3A800000u) return sign;
    int exp32 = (int)(ax >> 23) - 127;
    unsigned int mant = (ax & 0x7FFFFFu) | 0x800000u;
    int e = exp32 + 7;
    int shift = 20;
    if (e < 1) { shift += (1 - e); e = 0; }
    unsigned int lsb = (mant >> shift) & 1u;
    unsigned int rnd = (1u << (shift - 1)) - 1u + lsb;
    mant = (mant + rnd) >> shift;
    unsigned int out = (e > 0) ? (((unsigned int)(e - 1) << 3) + mant) : mant;
    if (out > 0x7Eu) out = 0x7Eu;
    return sign | out;
}

__global__ void init_amax_kernel(unsigned int* a) {
    if (threadIdx.x < 8) a[threadIdx.x] = 0u;
}

__global__ void amax_kernel(const float* __restrict__ x, size_t n,
                            unsigned int* __restrict__ slot) {
    size_t i      = (size_t)blockIdx.x * blockDim.x + threadIdx.x;
    size_t stride = (size_t)gridDim.x * blockDim.x;
    float m = 0.0f;
    for (; i < n; i += stride) m = fmaxf(m, fabsf(x[i]));
    for (int off = 16; off > 0; off >>= 1)
        m = fmaxf(m, __shfl_xor(m, off, 32));
    if ((threadIdx.x & 31) == 0) atomicMax(slot, __float_as_uint(m));
}

__global__ void quant_kernel(const float* __restrict__ x,
                             unsigned char* __restrict__ q, size_t n4,
                             const float* __restrict__ amax_slot) {
    float scale = 448.0f / fmaxf(amax_slot[0], 1e-12f);
    size_t i = (size_t)blockIdx.x * blockDim.x + threadIdx.x;
    if (i >= n4) return;
    float4 v = ((const float4*)x)[i];
    unsigned int p = f32_to_e4m3(v.x * scale)
                   | (f32_to_e4m3(v.y * scale) << 8)
                   | (f32_to_e4m3(v.z * scale) << 16)
                   | (f32_to_e4m3(v.w * scale) << 24);
    ((unsigned int*)q)[i] = p;
}

__global__ void quant_t_kernel(const float* __restrict__ w,
                               unsigned char* __restrict__ qt,
                               int F, int H,
                               const float* __restrict__ amax_slot) {
    float scale = 448.0f / fmaxf(amax_slot[0], 1e-12f);
    long long idx = (long long)blockIdx.x * blockDim.x + threadIdx.x;
    if (idx >= (long long)F * H) return;
    int f = (int)(idx / H);
    int h = (int)(idx % H);
    qt[(size_t)h * F + f] = (unsigned char)f32_to_e4m3(w[idx] * scale);
}

__device__ __forceinline__ v16i load_a_frag(const unsigned char* __restrict__ A,
                                            int lda, int row0, int k0, int lane) {
    int r = lane & 15, h = lane >> 4;
    const unsigned long long* p = (const unsigned long long*)
        (A + (size_t)(row0 + r) * lda + k0 + h * 16);
    v16i a;
#pragma unroll
    for (int j = 0; j < 8; ++j) {
        unsigned long long d = p[(j >> 1) * 4 + (j & 1)];
        a[2 * j]     = (int)(unsigned int)(d & 0xFFFFFFFFull);
        a[2 * j + 1] = (int)(unsigned int)(d >> 32);
    }
    return a;
}

__device__ __forceinline__ v16i lds_b_frag(const unsigned char* lb,
                                           int col0, int lane) {
    int c = lane & 15, h = lane >> 4;
    int t = (c >> 1) & 7;
    const unsigned char* p = lb + (unsigned)(col0 + c) * 128u;
    v16i b;
#pragma unroll
    for (int q = 0; q < 4; ++q) {
        int g = (h + 2 * q) ^ t;
        uint4 d = *(const uint4*)(p + g * 16);
        b[4 * q]     = (int)d.x;
        b[4 * q + 1] = (int)d.y;
        b[4 * q + 2] = (int)d.z;
        b[4 * q + 3] = (int)d.w;
    }
    return b;
}

__device__ __forceinline__ unsigned lds_chunk_off(int ch) {
    int col = ch >> 3, gr = ch & 7;
    return (unsigned)col * 128u + (unsigned)((gr ^ ((col >> 1) & 7)) << 4);
}

__global__ __launch_bounds__(256)
void gemm1_glu_kernel(const unsigned char* __restrict__ qx,
                      const unsigned char* __restrict__ qw1,
                      const unsigned char* __restrict__ qv1,
                      float* __restrict__ inter,
                      const float* __restrict__ amax,
                      unsigned int* __restrict__ amax_mid) {
    const int K = 4096, N = 10752;
    __shared__ unsigned char lds[2][2][16384];

    int tid  = threadIdx.x;
    int lane = tid & 31;
    int wave = tid >> 5;
    int m0   = blockIdx.y * 64 + (wave & 3) * 16;
    int nblk = blockIdx.x * 128;
    int n0   = (wave >> 2) * 64;

    v8f accg[4], accu[4];
#pragma unroll
    for (int t = 0; t < 4; ++t)
#pragma unroll
        for (int v = 0; v < 8; ++v) { accg[t][v] = 0.0f; accu[t][v] = 0.0f; }

    const unsigned char* bsrc[2] = { qw1, qv1 };
    const int nk = K / 128;

#if defined(HAVE_ASYNC_LDS)
#pragma unroll
    for (int m = 0; m < 2; ++m)
#pragma unroll
        for (int j = 0; j < 4; ++j) {
            int ch = tid * 4 + j;
            async_cp16(bsrc[m] + (size_t)(nblk + (ch >> 3)) * K + (ch & 7) * 16,
                       &lds[m][0][lds_chunk_off(ch)]);
        }
    for (int kk = 0; kk < nk; ++kk) {
        int buf = kk & 1;
        if (kk + 1 < nk) {
#pragma unroll
            for (int m = 0; m < 2; ++m)
#pragma unroll
                for (int j = 0; j < 4; ++j) {
                    int ch = tid * 4 + j;
                    async_cp16(bsrc[m] + (size_t)(nblk + (ch >> 3)) * K
                                       + (kk + 1) * 128 + (ch & 7) * 16,
                               &lds[m][buf ^ 1][lds_chunk_off(ch)]);
                }
            WAIT_ASYNC(8);
        } else {
            WAIT_ASYNC(0);
        }
        __syncthreads();
        v16i a = load_a_frag(qx, K, m0, kk * 128, lane);
#pragma unroll
        for (int t = 0; t < 4; ++t) {
            v16i bg = lds_b_frag(&lds[0][buf][0], n0 + t * 16, lane);
            accg[t] = __builtin_amdgcn_wmma_f32_16x16x128_fp8_fp8(
                a, bg, (short)0, accg[t], false, false);
            v16i bu = lds_b_frag(&lds[1][buf][0], n0 + t * 16, lane);
            accu[t] = __builtin_amdgcn_wmma_f32_16x16x128_fp8_fp8(
                a, bu, (short)0, accu[t], false, false);
        }
        __syncthreads();
    }
#else
    uint4 rg[2][4];
#pragma unroll
    for (int m = 0; m < 2; ++m)
#pragma unroll
        for (int j = 0; j < 4; ++j) {
            int ch = tid * 4 + j;
            rg[m][j] = *(const uint4*)(bsrc[m] + (size_t)(nblk + (ch >> 3)) * K
                                               + (ch & 7) * 16);
        }
#pragma unroll
    for (int m = 0; m < 2; ++m)
#pragma unroll
        for (int j = 0; j < 4; ++j)
            *(uint4*)&lds[m][0][lds_chunk_off(tid * 4 + j)] = rg[m][j];
    __syncthreads();

    for (int kk = 0; kk < nk; ++kk) {
        int buf = kk & 1;
        if (kk + 1 < nk) {
#pragma unroll
            for (int m = 0; m < 2; ++m)
#pragma unroll
                for (int j = 0; j < 4; ++j) {
                    int ch = tid * 4 + j;
                    rg[m][j] = *(const uint4*)(bsrc[m]
                                 + (size_t)(nblk + (ch >> 3)) * K
                                 + (kk + 1) * 128 + (ch & 7) * 16);
                }
        }
        v16i a = load_a_frag(qx, K, m0, kk * 128, lane);
#pragma unroll
        for (int t = 0; t < 4; ++t) {
            v16i bg = lds_b_frag(&lds[0][buf][0], n0 + t * 16, lane);
            accg[t] = __builtin_amdgcn_wmma_f32_16x16x128_fp8_fp8(
                a, bg, (short)0, accg[t], false, false);
            v16i bu = lds_b_frag(&lds[1][buf][0], n0 + t * 16, lane);
            accu[t] = __builtin_amdgcn_wmma_f32_16x16x128_fp8_fp8(
                a, bu, (short)0, accu[t], false, false);
        }
        __syncthreads();
        if (kk + 1 < nk) {
#pragma unroll
            for (int m = 0; m < 2; ++m)
#pragma unroll
                for (int j = 0; j < 4; ++j)
                    *(uint4*)&lds[m][buf ^ 1][lds_chunk_off(tid * 4 + j)] = rg[m][j];
        }
        __syncthreads();
    }
#endif

    const float inv = 1.0f / (448.0f * 448.0f);
    float s1 = amax[0] * amax[1] * inv;
    float s2 = amax[0] * amax[2] * inv;
    int r = lane & 15, h = lane >> 4;
    float lm = 0.0f;
#pragma unroll
    for (int t = 0; t < 4; ++t)
#pragma unroll
        for (int v = 0; v < 8; ++v) {
            float g   = accg[t][v] * s1;
            float u   = accu[t][v] * s2;
            float val = (g / (1.0f + __expf(-g))) * u;
            lm = fmaxf(lm, fabsf(val));
            inter[(size_t)(m0 + h * 8 + v) * N + (nblk + n0 + t * 16 + r)] = val;
        }
    for (int off = 16; off > 0; off >>= 1)
        lm = fmaxf(lm, __shfl_xor(lm, off, 32));
    if (lane == 0) atomicMax(amax_mid, __float_as_uint(lm));
}

__global__ __launch_bounds__(256)
void gemm2_kernel(const unsigned char* __restrict__ qmid,
                  const unsigned char* __restrict__ qw2t,
                  float* __restrict__ out,
                  const float* __restrict__ amax) {
    const int K = 10752, N = 4096;
    __shared__ unsigned char lds[2][16384];

    int tid  = threadIdx.x;
    int lane = tid & 31;
    int wave = tid >> 5;
    int m0   = blockIdx.y * 64 + (wave & 3) * 16;
    int nblk = blockIdx.x * 128;
    int n0   = (wave >> 2) * 64;

    v8f acc[4];
#pragma unroll
    for (int t = 0; t < 4; ++t)
#pragma unroll
        for (int v = 0; v < 8; ++v) acc[t][v] = 0.0f;

    const int nk = K / 128;

#if defined(HAVE_ASYNC_LDS)
#pragma unroll
    for (int j = 0; j < 4; ++j) {
        int ch = tid * 4 + j;
        async_cp16(qw2t + (size_t)(nblk + (ch >> 3)) * K + (ch & 7) * 16,
                   &lds[0][lds_chunk_off(ch)]);
    }
    for (int kk = 0; kk < nk; ++kk) {
        int buf = kk & 1;
        if (kk + 1 < nk) {
#pragma unroll
            for (int j = 0; j < 4; ++j) {
                int ch = tid * 4 + j;
                async_cp16(qw2t + (size_t)(nblk + (ch >> 3)) * K
                                + (kk + 1) * 128 + (ch & 7) * 16,
                           &lds[buf ^ 1][lds_chunk_off(ch)]);
            }
            WAIT_ASYNC(4);
        } else {
            WAIT_ASYNC(0);
        }
        __syncthreads();
        v16i a = load_a_frag(qmid, K, m0, kk * 128, lane);
#pragma unroll
        for (int t = 0; t < 4; ++t) {
            v16i b = lds_b_frag(&lds[buf][0], n0 + t * 16, lane);
            acc[t] = __builtin_amdgcn_wmma_f32_16x16x128_fp8_fp8(
                a, b, (short)0, acc[t], false, false);
        }
        __syncthreads();
    }
#else
    uint4 rg[4];
#pragma unroll
    for (int j = 0; j < 4; ++j) {
        int ch = tid * 4 + j;
        rg[j] = *(const uint4*)(qw2t + (size_t)(nblk + (ch >> 3)) * K
                                     + (ch & 7) * 16);
    }
#pragma unroll
    for (int j = 0; j < 4; ++j)
        *(uint4*)&lds[0][lds_chunk_off(tid * 4 + j)] = rg[j];
    __syncthreads();

    for (int kk = 0; kk < nk; ++kk) {
        int buf = kk & 1;
        if (kk + 1 < nk) {
#pragma unroll
            for (int j = 0; j < 4; ++j) {
                int ch = tid * 4 + j;
                rg[j] = *(const uint4*)(qw2t + (size_t)(nblk + (ch >> 3)) * K
                                             + (kk + 1) * 128 + (ch & 7) * 16);
            }
        }
        v16i a = load_a_frag(qmid, K, m0, kk * 128, lane);
#pragma unroll
        for (int t = 0; t < 4; ++t) {
            v16i b = lds_b_frag(&lds[buf][0], n0 + t * 16, lane);
            acc[t] = __builtin_amdgcn_wmma_f32_16x16x128_fp8_fp8(
                a, b, (short)0, acc[t], false, false);
        }
        __syncthreads();
        if (kk + 1 < nk) {
#pragma unroll
            for (int j = 0; j < 4; ++j)
                *(uint4*)&lds[buf ^ 1][lds_chunk_off(tid * 4 + j)] = rg[j];
        }
        __syncthreads();
    }
#endif

    float s = amax[4] * amax[3] * (1.0f / (448.0f * 448.0f));
    int r = lane & 15, h = lane >> 4;
#pragma unroll
    for (int t = 0; t < 4; ++t)
#pragma unroll
        for (int v = 0; v < 8; ++v)
            out[(size_t)(m0 + h * 8 + v) * N + (nblk + n0 + t * 16 + r)] = acc[t][v] * s;
}

extern "C" void kernel_launch(void* const* d_in, const int* in_sizes, int n_in,
                              void* d_out, int out_size, void* d_ws, size_t ws_size,
                              hipStream_t stream) {
    (void)in_sizes; (void)n_in; (void)out_size; (void)ws_size;
    const int TOK = 4096, HID = 4096, FFN = 10752;

    const float* x  = (const float*)d_in[0];
    const float* w1 = (const float*)d_in[1];
    const float* v1 = (const float*)d_in[2];
    const float* w2 = (const float*)d_in[3];
    float* out = (float*)d_out;

    char* ws = (char*)d_ws;
    size_t off = 0;
    float*        amax  = (float*)ws;                         off += 256;
    unsigned char* qx   = (unsigned char*)(ws + off);         off += (size_t)TOK * HID;
    unsigned char* qw1  = (unsigned char*)(ws + off);         off += (size_t)FFN * HID;
    unsigned char* qv1  = (unsigned char*)(ws + off);         off += (size_t)FFN * HID;
    unsigned char* qw2t = (unsigned char*)(ws + off);         off += (size_t)HID * FFN;
    unsigned char* qmid = (unsigned char*)(ws + off);         off += (size_t)TOK * FFN;
    float*        inter = (float*)(ws + off);

    unsigned int* amax_u = (unsigned int*)amax;

    init_amax_kernel<<<1, 32, 0, stream>>>(amax_u);

    amax_kernel<<<1024, 256, 0, stream>>>(x,  (size_t)TOK * HID, amax_u + 0);
    amax_kernel<<<2048, 256, 0, stream>>>(w1, (size_t)FFN * HID, amax_u + 1);
    amax_kernel<<<2048, 256, 0, stream>>>(v1, (size_t)FFN * HID, amax_u + 2);
    amax_kernel<<<2048, 256, 0, stream>>>(w2, (size_t)FFN * HID, amax_u + 3);

    size_t n4x = (size_t)TOK * HID / 4;
    size_t n4w = (size_t)FFN * HID / 4;
    quant_kernel<<<(unsigned)((n4x + 255) / 256), 256, 0, stream>>>(x,  qx,  n4x, amax + 0);
    quant_kernel<<<(unsigned)((n4w + 255) / 256), 256, 0, stream>>>(w1, qw1, n4w, amax + 1);
    quant_kernel<<<(unsigned)((n4w + 255) / 256), 256, 0, stream>>>(v1, qv1, n4w, amax + 2);

    long long nw2 = (long long)FFN * HID;
    quant_t_kernel<<<(unsigned)((nw2 + 255) / 256), 256, 0, stream>>>(w2, qw2t, FFN, HID, amax + 3);

    dim3 g1(FFN / 128, TOK / 64);
    gemm1_glu_kernel<<<g1, 256, 0, stream>>>(qx, qw1, qv1, inter, amax, amax_u + 4);

    size_t n4m = (size_t)TOK * FFN / 4;
    quant_kernel<<<(unsigned)((n4m + 255) / 256), 256, 0, stream>>>(inter, qmid, n4m, amax + 4);

    dim3 g2(HID / 128, TOK / 64);
    gemm2_kernel<<<g2, 256, 0, stream>>>(qmid, qw2t, out, amax);
}
